// RNN_21406117003940
// MI455X (gfx1250) — matrix operand order repaired and checked by reading, not compiled, not run on hardware
//
#include <hip/hip_runtime.h>
#include <hip/hip_bf16.h>

#define SEQ_LEN 1024
#define BATCH   256
#define EMBED   512
#define HIDDEN  1024

#define LDA1 520   // 512 + 8 pad: 16B-aligned rows, 4-bank rotation per row
#define LDA2 1032  // 1024 + 8 pad: 16B-aligned rows, 4-bank rotation per row

typedef __attribute__((ext_vector_type(16))) __bf16 v16bf;
typedef __attribute__((ext_vector_type(8)))  float  v8f;

union FragU { uint4 q[2]; v16bf v; };

// A-matrix fragment (16x32 bf16): lane L holds row M=L%16;
// lanes 0-15 take K = 0..15, lanes 16-31 take K = 16..31.
// Caller pre-offsets p by (lane>=16 ? 16 : 0) elements; we read +0 and +8.
__device__ __forceinline__ v16bf ldfragA(const __hip_bfloat16* p) {
  FragU u;
  u.q[0] = *(const uint4*)(p);
  u.q[1] = *(const uint4*)(p + 8);
  return u.v;
}

// B-matrix fragment (32x16 bf16): lane L holds column N=L%16 with 16
// consecutive K values; lanes 16-31 take K+16. Caller pre-offsets by
// (lane>=16 ? 16 : 0); we read two contiguous 16B chunks.
__device__ __forceinline__ v16bf ldfragB(const __hip_bfloat16* p) {
  FragU u;
  u.q[0] = *(const uint4*)(p);
  u.q[1] = *(const uint4*)(p + 8);
  return u.v;
}

#define WMMA_BF16(a, b, c) \
  __builtin_amdgcn_wmma_f32_16x16x32_bf16(false, (a), false, (b), (short)0, (c), false, false)

// ---- gfx1250 async global->LDS DMA (ASYNCcnt-tracked), with fallback ------
// Probe-confirmed param 0 type (from hipcc diagnostic): GCC-vector int4 in
// AS(1), i.e. (global int4*, lds int4*, imm offset, imm cpol).
#if __has_builtin(__builtin_amdgcn_global_load_async_to_lds_b128)
#define HAVE_ASYNC_LDS 1
typedef int v4i_gcc __attribute__((vector_size(16)));
__device__ __forceinline__ void async_copy16(const void* g, void* l) {
  __builtin_amdgcn_global_load_async_to_lds_b128(
      (__attribute__((address_space(1))) v4i_gcc*)g,
      (__attribute__((address_space(3))) v4i_gcc*)l, 0, 0);
}
#if __has_builtin(__builtin_amdgcn_s_wait_asynccnt)
#define WAIT_ASYNC() __builtin_amdgcn_s_wait_asynccnt(0)
#else
#define WAIT_ASYNC() asm volatile("s_wait_asynccnt 0x0" ::: "memory")
#endif
#else
#define HAVE_ASYNC_LDS 0
#define WAIT_ASYNC() ((void)0)
#endif

// ---------------------------------------------------------------------------
// Prep: split weights into bf16 hi/lo pairs, fold the two biases together.
// ---------------------------------------------------------------------------
__global__ __launch_bounds__(256)
void rnn_prep_kernel(const float* __restrict__ W_ih, const float* __restrict__ b_ih,
                     const float* __restrict__ W_hh, const float* __restrict__ b_hh,
                     __hip_bfloat16* __restrict__ wih_hi, __hip_bfloat16* __restrict__ wih_lo,
                     __hip_bfloat16* __restrict__ whh_hi, __hip_bfloat16* __restrict__ whh_lo,
                     float* __restrict__ bias) {
  const int i = blockIdx.x * blockDim.x + threadIdx.x;
  if (i < HIDDEN) bias[i] = b_ih[i] + b_hh[i];
  if (i < HIDDEN * EMBED) {
    float x = W_ih[i];
    __hip_bfloat16 hi = __float2bfloat16(x);
    wih_hi[i] = hi;
    wih_lo[i] = __float2bfloat16(x - __bfloat162float(hi));
  }
  const int j = i - HIDDEN * EMBED;
  if (j >= 0 && j < HIDDEN * HIDDEN) {
    float x = W_hh[j];
    __hip_bfloat16 hi = __float2bfloat16(x);
    whh_hi[j] = hi;
    whh_lo[j] = __float2bfloat16(x - __bfloat162float(hi));
  }
}

__global__ __launch_bounds__(256)
void rnn_zero_kernel(__hip_bfloat16* __restrict__ a, __hip_bfloat16* __restrict__ b) {
  const int i = blockIdx.x * blockDim.x + threadIdx.x;
  const __hip_bfloat16 z = __float2bfloat16(0.0f);
  a[i] = z;
  b[i] = z;
}

// ---------------------------------------------------------------------------
// One recurrence step: h_t = tanh(W_ih @ emb[x_t] + W_hh @ h_{t-1} + bias)
// Block = 256 threads (8 waves). Block tile: M=16 batch rows x N=128 hidden.
// Each wave owns one 16x16 tile, K = 512 (embedding) + 1024 (recurrent),
// computed as split-bf16: acc += aH*bH + aH*bL + aL*bH  (~f32 accuracy).
// A operands are staged in LDS once per block (async DMA for the h panel,
// overlapped with the embedding gather+split); B streams from L2.
// ---------------------------------------------------------------------------
__global__ __launch_bounds__(256)
void rnn_step_kernel(const int* __restrict__ x_seq,
                     const float* __restrict__ emb,
                     const __hip_bfloat16* __restrict__ wih_hi,
                     const __hip_bfloat16* __restrict__ wih_lo,
                     const __hip_bfloat16* __restrict__ whh_hi,
                     const __hip_bfloat16* __restrict__ whh_lo,
                     const float* __restrict__ bias,
                     const __hip_bfloat16* __restrict__ hcur_hi,
                     const __hip_bfloat16* __restrict__ hcur_lo,
                     __hip_bfloat16* __restrict__ hnxt_hi,
                     __hip_bfloat16* __restrict__ hnxt_lo,
                     float* __restrict__ out,
                     int t) {
  __shared__ __hip_bfloat16 s1_hi[16 * LDA1];  // embedded inputs, hi
  __shared__ __hip_bfloat16 s1_lo[16 * LDA1];  // embedded inputs, lo
  __shared__ __hip_bfloat16 s2_hi[16 * LDA2];  // h_{t-1} rows, hi
  __shared__ __hip_bfloat16 s2_lo[16 * LDA2];  // h_{t-1} rows, lo

  const int tid   = threadIdx.x;
  const int mbase = blockIdx.x * 16;
  const int nbase = blockIdx.y * 128 + (tid >> 5) * 16;

  // ---- Kick off async DMA of the h_{t-1} panel (K = 512..1535 A operand).
  // 16 rows x 1024 bf16 (hi+lo); 256 threads x 8 x 2 b128 transfers.
  {
    const int row = tid >> 4;          // 0..15
    const int c0  = (tid & 15) * 64;   // 0..960
    const __hip_bfloat16* gh = hcur_hi + (size_t)(mbase + row) * HIDDEN + c0;
    const __hip_bfloat16* gl = hcur_lo + (size_t)(mbase + row) * HIDDEN + c0;
    __hip_bfloat16* dh = s2_hi + row * LDA2 + c0;
    __hip_bfloat16* dl = s2_lo + row * LDA2 + c0;
#if HAVE_ASYNC_LDS
#pragma unroll
    for (int q = 0; q < 8; ++q) {
      async_copy16(gh + q * 8, dh + q * 8);
      async_copy16(gl + q * 8, dl + q * 8);
    }
#else
#pragma unroll
    for (int q = 0; q < 8; ++q) {
      *(uint4*)(dh + q * 8) = *(const uint4*)(gh + q * 8);
      *(uint4*)(dl + q * 8) = *(const uint4*)(gl + q * 8);
    }
#endif
  }

  // ---- Meanwhile: gather embedding rows (K = 0..511) and split into LDS.
  {
    const int row = tid >> 4;          // 0..15
    const int c0  = (tid & 15) * 32;   // 0..480
    const int token = x_seq[t * BATCH + mbase + row];
    const float4* src = (const float4*)(emb + (size_t)token * EMBED + c0);
    __hip_bfloat16* dh = s1_hi + row * LDA1 + c0;
    __hip_bfloat16* dl = s1_lo + row * LDA1 + c0;
#pragma unroll
    for (int v = 0; v < 8; ++v) {
      float4 f = src[v];
      float vals[4] = {f.x, f.y, f.z, f.w};
#pragma unroll
      for (int e = 0; e < 4; ++e) {
        float x = vals[e];
        __hip_bfloat16 hi = __float2bfloat16(x);
        dh[v * 4 + e] = hi;
        dl[v * 4 + e] = __float2bfloat16(x - __bfloat162float(hi));
      }
    }
  }
  WAIT_ASYNC();
  __syncthreads();

  const int lane = tid & 31;
  const int nl   = lane & 15;   // column within tile / A row within tile
  const int hi16 = lane >> 4;   // lane-group selector

  v8f acc = {};

  // ---- Part 1: input projection, K = 0..511 (A from LDS, B = W_ih) ----
  {
    const __hip_bfloat16* aH = s1_hi + nl * LDA1 + hi16 * 16;
    const __hip_bfloat16* aL = s1_lo + nl * LDA1 + hi16 * 16;
    const __hip_bfloat16* bH = wih_hi + (size_t)(nbase + nl) * EMBED + hi16 * 16;
    const __hip_bfloat16* bL = wih_lo + (size_t)(nbase + nl) * EMBED + hi16 * 16;
#pragma unroll 4
    for (int kk = 0; kk < EMBED / 32; ++kk) {
      v16bf ah = ldfragA(aH + kk * 32);
      v16bf al = ldfragA(aL + kk * 32);
      v16bf bh = ldfragB(bH + kk * 32);
      v16bf bl = ldfragB(bL + kk * 32);
      acc = WMMA_BF16(ah, bh, acc);
      acc = WMMA_BF16(ah, bl, acc);
      acc = WMMA_BF16(al, bh, acc);
    }
  }

  // ---- Part 2: recurrent projection, K over h_{t-1} (A from LDS, B = W_hh) ----
  {
    const __hip_bfloat16* aH = s2_hi + nl * LDA2 + hi16 * 16;
    const __hip_bfloat16* aL = s2_lo + nl * LDA2 + hi16 * 16;
    const __hip_bfloat16* bH = whh_hi + (size_t)(nbase + nl) * HIDDEN + hi16 * 16;
    const __hip_bfloat16* bL = whh_lo + (size_t)(nbase + nl) * HIDDEN + hi16 * 16;
#pragma unroll 4
    for (int kk = 0; kk < HIDDEN / 32; ++kk) {
      v16bf ah = ldfragA(aH + kk * 32);
      v16bf al = ldfragA(aL + kk * 32);
      v16bf bh = ldfragB(bH + kk * 32);
      v16bf bl = ldfragB(bL + kk * 32);
      acc = WMMA_BF16(ah, bh, acc);
      acc = WMMA_BF16(ah, bl, acc);
      acc = WMMA_BF16(al, bh, acc);
    }
  }

  // ---- Epilogue: bias + tanh; write f32 output and split-bf16 h_next ----
  // C/D layout: VGPR j, lanes 0-15 -> (M=j, N=lane); lanes 16-31 -> (M=j+8).
  {
    const int n  = nbase + nl;
    const float bn = bias[n];
    const int m0 = mbase + hi16 * 8;
#pragma unroll
    for (int j = 0; j < 8; ++j) {
      const int m = m0 + j;
      float v = tanhf(acc[j] + bn);
      out[((size_t)t * BATCH + m) * HIDDEN + n] = v;
      __hip_bfloat16 hi = __float2bfloat16(v);
      hnxt_hi[(size_t)m * HIDDEN + n] = hi;
      hnxt_lo[(size_t)m * HIDDEN + n] = __float2bfloat16(v - __bfloat162float(hi));
    }
  }
}

// ---------------------------------------------------------------------------
extern "C" void kernel_launch(void* const* d_in, const int* in_sizes, int n_in,
                              void* d_out, int out_size, void* d_ws, size_t ws_size,
                              hipStream_t stream) {
  const int*   x_seq = (const int*)  d_in[0];
  const float* emb   = (const float*)d_in[1];
  const float* W_ih  = (const float*)d_in[2];
  const float* b_ih  = (const float*)d_in[3];
  const float* W_hh  = (const float*)d_in[4];
  const float* b_hh  = (const float*)d_in[5];
  float* out = (float*)d_out;

  char* ws = (char*)d_ws;
  size_t off = 0;
  auto take = [&](size_t bytes) -> char* {
    char* p = ws + off;
    off += (bytes + 255) & ~(size_t)255;
    return p;
  };
  __hip_bfloat16* wih_hi = (__hip_bfloat16*)take((size_t)HIDDEN * EMBED * 2);
  __hip_bfloat16* wih_lo = (__hip_bfloat16*)take((size_t)HIDDEN * EMBED * 2);
  __hip_bfloat16* whh_hi = (__hip_bfloat16*)take((size_t)HIDDEN * HIDDEN * 2);
  __hip_bfloat16* whh_lo = (__hip_bfloat16*)take((size_t)HIDDEN * HIDDEN * 2);
  float*          bias   = (float*)take((size_t)HIDDEN * 4);
  __hip_bfloat16* h_hi0  = (__hip_bfloat16*)take((size_t)BATCH * HIDDEN * 2);
  __hip_bfloat16* h_lo0  = (__hip_bfloat16*)take((size_t)BATCH * HIDDEN * 2);
  __hip_bfloat16* h_hi1  = (__hip_bfloat16*)take((size_t)BATCH * HIDDEN * 2);
  __hip_bfloat16* h_lo1  = (__hip_bfloat16*)take((size_t)BATCH * HIDDEN * 2);

  // Split weights + fold bias (covers HIDDEN*EMBED + HIDDEN*HIDDEN elements).
  {
    const int total = HIDDEN * EMBED + HIDDEN * HIDDEN;
    rnn_prep_kernel<<<(total + 255) / 256, 256, 0, stream>>>(
        W_ih, b_ih, W_hh, b_hh, wih_hi, wih_lo, whh_hi, whh_lo, bias);
  }
  // h_{-1} = 0 (ping buffer).
  rnn_zero_kernel<<<(BATCH * HIDDEN) / 256, 256, 0, stream>>>(h_hi0, h_lo0);

  const dim3 grid(BATCH / 16, HIDDEN / 128);
  for (int t = 0; t < SEQ_LEN; ++t) {
    const __hip_bfloat16* cur_hi = (t & 1) ? h_hi1 : h_hi0;
    const __hip_bfloat16* cur_lo = (t & 1) ? h_lo1 : h_lo0;
    __hip_bfloat16* nxt_hi = (t & 1) ? h_hi0 : h_hi1;
    __hip_bfloat16* nxt_lo = (t & 1) ? h_lo0 : h_lo1;
    rnn_step_kernel<<<grid, 256, 0, stream>>>(
        x_seq, emb, wih_hi, wih_lo, whh_hi, whh_lo, bias,
        cur_hi, cur_lo, nxt_hi, nxt_lo, out, t);
  }
}
